// MultiHeadAttention_2860448219502
// MI455X (gfx1250) — matrix operand order repaired and checked by reading, not compiled, not run on hardware
//
#include <hip/hip_runtime.h>

// =====================================================================
// MI455X (gfx1250) fused MHA: bf16 WMMA + TDM async tensor loads.
//   1) Q = (query@Wq+bq)*0.125   -> bf16 [B,H,T,64]   (mha_proj_wmma<0>)
//      K = (value@Wk+bk)         -> bf16 [B,H,T,64]   (mha_proj_wmma<0>)
//      Vt= (value@Wv+bv)^T       -> bf16 [B,H,64,T]   (mha_proj_wmma<1>)
//   2) flash attention, 4 waves/block (64 q-rows), K/V tiles streamed
//      into LDS by the Tensor Data Mover (double-buffered, TENSORcnt)
//   3) out = ctx@Wo + bo         -> f32  [B*T,1024]   (mha_proj_wmma<2>)
// Workspace: 3*8MB bf16 + 16MB f32 = 40 MB (fits in d_ws and in L2).
// =====================================================================

typedef __attribute__((ext_vector_type(16))) __bf16        v16bf;
typedef __attribute__((ext_vector_type(8)))  float         v8f;
typedef __attribute__((ext_vector_type(4)))  unsigned int  u32x4;
typedef __attribute__((ext_vector_type(8)))  int           i32x8;
typedef __attribute__((ext_vector_type(4)))  int           i32x4;

#define DEV __device__ __forceinline__

union FragU {
  v16bf v;
  uint4 q[2];
};

// A-fragment (16x32 bf16, MxK): lane holds row m=lane&15; K elements are
// two contiguous 8-runs at [kb..kb+7] and [16+kb..16+kb+7], kb=(lane>>4)*8.
DEV v16bf load_fragA(const __bf16* row, int kb) {
  FragU f;
  f.q[0] = *reinterpret_cast<const uint4*>(row + kb);
  f.q[1] = *reinterpret_cast<const uint4*>(row + 16 + kb);
  return f.v;
}

// B-fragment (32x16 bf16, KxN): lane holds col n=lane&15; K elements are
// two 8-runs at [kb..kb+7] and [16+kb..16+kb+7], kb=kb2/2=(lane>>4)*8.
// `row` points at the n-th row of B^T.
DEV v16bf load_fragB(const __bf16* row, int kb2) {
  FragU f;
  f.q[0] = *reinterpret_cast<const uint4*>(row + (kb2 >> 1));
  f.q[1] = *reinterpret_cast<const uint4*>(row + 16 + (kb2 >> 1));
  return f.v;
}

DEV v8f wmma_bf16(v16bf a, v16bf b, v8f c) {
  return __builtin_amdgcn_wmma_f32_16x16x32_bf16(
      false, a, false, b, (short)0, c, false, false);
}

DEV float half_max(float x) {
  x = fmaxf(x, __shfl_xor(x, 1));
  x = fmaxf(x, __shfl_xor(x, 2));
  x = fmaxf(x, __shfl_xor(x, 4));
  x = fmaxf(x, __shfl_xor(x, 8));
  return x;
}
DEV float half_sum(float x) {
  x += __shfl_xor(x, 1);
  x += __shfl_xor(x, 2);
  x += __shfl_xor(x, 4);
  x += __shfl_xor(x, 8);
  return x;
}

// LDS byte address of a generic pointer to __shared__ data: the LDS
// aperture maps generic->LDS by truncating to addr[31:0] (ISA 10.2).
DEV unsigned lds_byte_off(const void* p) {
  return (unsigned)(unsigned long long)(uintptr_t)p;
}

// ---------------------------------------------------------------------
// TDM: issue one 2-D tensor_load_to_lds.  D# per CDNA5 ISA sec.8:
//  group0: dw0 {count=1}, dw1 lds_addr, dw2 ga[31:0],
//          dw3 ga[56:32] | type=2<<30
//  group1: dw0 data_size=1(2B), dw1 tensor_dim0<<16, dw2 dim0_hi|dim1<<16,
//          dw3 dim1_hi|tile0<<16, dw4 tile1, dw5 stride0, dw6/7 hi bits
//  groups 2/3 (+extra arg): zero (<=2-D tensor, no iterate/gather).
//  This toolchain declares the 6-arg builtin:
//  (u32x4, i32x8, i32x4, i32x4, i32x8, i32 cpol).
// ---------------------------------------------------------------------
DEV void tdm_load_2d(unsigned lds_off, unsigned long long ga,
                     int tdim0, int tdim1, int tile0, int tile1, int stride0) {
  u32x4 g0;
  g0[0] = 1u;                                   // count=1, no gather
  g0[1] = lds_off;                              // lds_addr (bytes)
  g0[2] = (unsigned)ga;                         // global_addr[31:0]
  g0[3] = (unsigned)((ga >> 32) & 0x01FFFFFFu) | (2u << 30);  // [56:32]|type=2
  i32x8 g1;
  g1[0] = 0x00010000;                           // data_size = 2 bytes
  g1[1] = (tdim0 & 0xFFFF) << 16;               // tensor_dim0[15:0]
  g1[2] = ((tdim0 >> 16) & 0xFFFF) | ((tdim1 & 0xFFFF) << 16);
  g1[3] = ((tdim1 >> 16) & 0xFFFF) | ((tile0 & 0xFFFF) << 16);
  g1[4] = (tile1 & 0xFFFF);                     // tile_dim1, tile_dim2=0
  g1[5] = stride0;                              // tensor_dim0_stride[31:0]
  g1[6] = 0;
  g1[7] = 0;
  i32x4 z4 = {0, 0, 0, 0};
  i32x8 z8 = {0, 0, 0, 0, 0, 0, 0, 0};
  __builtin_amdgcn_tensor_load_to_lds(g0, g1, z4, z4, z8, 0);
}

// ---------------------------------------------------------------------
// Generic 4096x1024x1024 GEMM (+bias, *scale).  One wave per 16x16 C
// tile, 32-step K loop of WMMAs; weight tile staged transposed in LDS so
// B-fragment gathers are ds_load_b128.
// MODE 0: bf16 out [B,H,T,64] | MODE 1: bf16 out [B,H,64,T] | MODE 2: f32.
// ---------------------------------------------------------------------
template <int MODE>
__global__ void __launch_bounds__(32)
mha_proj_wmma(const float* __restrict__ A, const float* __restrict__ W,
              const float* __restrict__ bias, float scale,
              void* __restrict__ outp) {
  __shared__ __align__(16) __bf16 Wt[16 * 32];  // [n_local][k_local]
  const int lane = threadIdx.x;
  const int m0   = lane & 15;
  const int kb   = (lane >> 4) << 3;
  const int kb2  = (lane >> 4) << 4;
  const int ibase = blockIdx.x << 4;
  const int nbase = blockIdx.y << 4;

  const float* arow = A + (size_t)(ibase + m0) * 1024;
  v8f acc = {};

  for (int k0 = 0; k0 < 1024; k0 += 32) {
    if (k0 + 32 < 1024)
      __builtin_prefetch(&W[(size_t)(k0 + 32) * 1024 + nbase], 0, 0);
    const int kshift = lane >> 4;
#pragma unroll
    for (int i = 0; i < 16; ++i) {
      const int kk = (i << 1) + kshift;
      Wt[(m0 << 5) + kk] = (__bf16)W[(size_t)(k0 + kk) * 1024 + nbase + m0];
    }
    __syncthreads();
    v16bf afrag;
#pragma unroll
    for (int i = 0; i < 8; ++i) {
      afrag[i]     = (__bf16)arow[k0 + kb + i];
      afrag[i + 8] = (__bf16)arow[k0 + 16 + kb + i];
    }
    const v16bf bfrag = load_fragB(&Wt[m0 << 5], kb2);
    acc = wmma_bf16(afrag, bfrag, acc);
    __syncthreads();
  }

  const int   n  = nbase + m0;
  const float bv = bias[n];
#pragma unroll
  for (int r = 0; r < 8; ++r) {
    const int   m   = r + ((lane >> 4) << 3);
    const int   row = ibase + m;
    const float val = (acc[r] + bv) * scale;
    if (MODE == 0) {
      const int bq = row >> 11, t = row & 2047;
      const int h = n >> 6, dk = n & 63;
      ((__bf16*)outp)[((((size_t)bq << 4) + h) * 2048 + t) * 64 + dk] =
          (__bf16)val;
    } else if (MODE == 1) {
      const int bq = row >> 11, t = row & 2047;
      const int h = n >> 6, dv = n & 63;
      ((__bf16*)outp)[((((size_t)bq << 4) + h) * 64 + dv) * 2048 + t] =
          (__bf16)val;
    } else {
      ((float*)outp)[(size_t)row * 1024 + n] = val;
    }
  }
}

// ---------------------------------------------------------------------
// Flash attention: 128 threads = 4 waves, 64 query rows per block.
// K tile [32 s][64 dk] and V tile [64 dv][32 s] are TDM-streamed into
// LDS (double-buffered); each tile is consumed by all 4 waves.
// Per 32-wide S tile: 4 WMMAs scores, online softmax, P through per-wave
// LDS (C-frag -> A-frag relayout), 4 WMMAs P*V.
// ---------------------------------------------------------------------
__global__ void __launch_bounds__(128)
mha_flash_wmma(const __bf16* __restrict__ Qb, const __bf16* __restrict__ Kb,
               const __bf16* __restrict__ Vt, const int* __restrict__ mask,
               float* __restrict__ ctx) {
  __shared__ __align__(16) __bf16 Klds[2][32 * 64];  // [buf][s][dk]
  __shared__ __align__(16) __bf16 Vlds[2][64 * 32];  // [buf][dv][s]
  __shared__ __align__(16) __bf16 Plds[4][16 * 32];  // per-wave P tile

  const int tid  = threadIdx.x;
  const int lane = tid & 31;
  const int wv   = tid >> 5;
  const int m0   = lane & 15;
  const int kb   = (lane >> 4) << 3;
  const int kb2  = (lane >> 4) << 4;
  const int ibase = (blockIdx.x << 6) + (wv << 4);
  const int h = blockIdx.y, b = blockIdx.z;

  const size_t head = (((size_t)b << 4) + h) * (size_t)(2048 * 64);
  const __bf16* Qh = Qb + head;
  const __bf16* Kh = Kb + head;  // [2048][64]
  const __bf16* Vh = Vt + head;  // [64][2048]
  const int* maskrow = mask + ((size_t)b * 2048 + ibase) * 2048;

  const unsigned long long kg = (unsigned long long)(uintptr_t)Kh;
  const unsigned long long vg = (unsigned long long)(uintptr_t)Vh;
  const unsigned ldsK[2] = {lds_byte_off(&Klds[0][0]), lds_byte_off(&Klds[1][0])};
  const unsigned ldsV[2] = {lds_byte_off(&Vlds[0][0]), lds_byte_off(&Vlds[1][0])};

  // Q tile fragments, register-resident for the whole S loop.
  const v16bf qA0 = load_fragA(Qh + (size_t)(ibase + m0) * 64, kb);
  const v16bf qA1 = load_fragA(Qh + (size_t)(ibase + m0) * 64 + 32, kb);

  v8f o0 = {}, o1 = {}, o2 = {}, o3 = {};
  float mrun[8], lrun[8];
#pragma unroll
  for (int r = 0; r < 8; ++r) { mrun[r] = -3.0e38f; lrun[r] = 0.0f; }

  // Prologue: TDM-stage tile j=0 into buffer 0 (wave 0 drives the TDM).
  if (wv == 0) {
    tdm_load_2d(ldsK[0], kg, 64, 2048, 64, 32, 64);          // K rows 0..31
    tdm_load_2d(ldsV[0], vg, 2048, 64, 32, 64, 2048);        // V cols 0..31
  }

  for (int j = 0; j < 2048; j += 32) {
    const int cur = (j >> 5) & 1;
    if (wv == 0) __builtin_amdgcn_s_wait_tensorcnt(0);  // tile j landed
    __syncthreads();  // all waves: tile j visible; prev buf free for reuse
    if (wv == 0 && (j + 32) < 2048) {                   // prefetch tile j+32
      const int nxt = cur ^ 1;
      tdm_load_2d(ldsK[nxt], kg + (size_t)(j + 32) * 128, 64, 2048, 64, 32, 64);
      tdm_load_2d(ldsV[nxt], vg + (size_t)(j + 32) * 2, 2048, 64, 32, 64, 2048);
    }

    const __bf16* Kc = &Klds[cur][0];
    const __bf16* Vc = &Vlds[cur][0];
    const __bf16* k0row = Kc + (size_t)m0 * 64;         // n-tile 0: s=j+n
    const __bf16* k1row = Kc + (size_t)(16 + m0) * 64;  // n-tile 1
    v8f s0 = {}, s1 = {};
    s0 = wmma_bf16(qA0, load_fragB(k0row, kb2), s0);
    s0 = wmma_bf16(qA1, load_fragB(k0row + 32, kb2), s0);
    s1 = wmma_bf16(qA0, load_fragB(k1row, kb2), s1);
    s1 = wmma_bf16(qA1, load_fragB(k1row + 32, kb2), s1);

#pragma unroll
    for (int r = 0; r < 8; ++r) {
      const int   m = r + ((lane >> 4) << 3);
      const float ad0 =
          (float)(1 - maskrow[(size_t)m * 2048 + j + m0]) * -10000.0f;
      const float ad1 =
          (float)(1 - maskrow[(size_t)m * 2048 + j + 16 + m0]) * -10000.0f;
      const float v0 = s0[r] + ad0;
      const float v1 = s1[r] + ad1;
      const float rm    = half_max(fmaxf(v0, v1));
      const float mnew  = fmaxf(mrun[r], rm);
      const float alpha = __expf(mrun[r] - mnew);
      mrun[r] = mnew;
      const float p0 = __expf(v0 - mnew);
      const float p1 = __expf(v1 - mnew);
      lrun[r] = lrun[r] * alpha + half_sum(p0 + p1);
      o0[r] *= alpha; o1[r] *= alpha; o2[r] *= alpha; o3[r] *= alpha;
      Plds[wv][(m << 5) + m0]      = (__bf16)p0;
      Plds[wv][(m << 5) + 16 + m0] = (__bf16)p1;
    }
    __syncthreads();  // P visible (and orders P stores vs. b128 reload)

    const v16bf pA = load_fragA(&Plds[wv][m0 << 5], kb);
    o0 = wmma_bf16(pA, load_fragB(Vc + (size_t)( 0 + m0) * 32, kb2), o0);
    o1 = wmma_bf16(pA, load_fragB(Vc + (size_t)(16 + m0) * 32, kb2), o1);
    o2 = wmma_bf16(pA, load_fragB(Vc + (size_t)(32 + m0) * 32, kb2), o2);
    o3 = wmma_bf16(pA, load_fragB(Vc + (size_t)(48 + m0) * 32, kb2), o3);
  }

  float* crow = ctx + ((size_t)b * 2048 + ibase) * 1024 + (h << 6);
#pragma unroll
  for (int r = 0; r < 8; ++r) {
    const int   m   = r + ((lane >> 4) << 3);
    const float inv = 1.0f / lrun[r];
    crow[(size_t)m * 1024 +  0 + m0] = o0[r] * inv;
    crow[(size_t)m * 1024 + 16 + m0] = o1[r] * inv;
    crow[(size_t)m * 1024 + 32 + m0] = o2[r] * inv;
    crow[(size_t)m * 1024 + 48 + m0] = o3[r] * inv;
  }
}

// ---------------------------------------------------------------------
extern "C" void kernel_launch(void* const* d_in, const int* in_sizes, int n_in,
                              void* d_out, int out_size, void* d_ws,
                              size_t ws_size, hipStream_t stream) {
  (void)in_sizes; (void)n_in; (void)out_size; (void)ws_size;
  const float* query = (const float*)d_in[0];
  const float* value = (const float*)d_in[1];
  const float* Wq    = (const float*)d_in[2];
  const float* bq    = (const float*)d_in[3];
  const float* Wk    = (const float*)d_in[4];
  const float* bk    = (const float*)d_in[5];
  const float* Wv    = (const float*)d_in[6];
  const float* bv    = (const float*)d_in[7];
  const float* Wo    = (const float*)d_in[8];   // [H*DV, D] row-major flat
  const float* bo    = (const float*)d_in[9];
  const int*   amask = (const int*)d_in[10];
  float* out = (float*)d_out;

  const size_t HEAD_ELEMS = (size_t)2 * 16 * 2048 * 64;  // 4M bf16 each
  __bf16* Qb = (__bf16*)d_ws;
  __bf16* Kb = Qb + HEAD_ELEMS;
  __bf16* Vt = Kb + HEAD_ELEMS;
  float*  ctx = (float*)(Vt + HEAD_ELEMS);

  const dim3 gproj(256, 64), blk32(32);
  mha_proj_wmma<0><<<gproj, blk32, 0, stream>>>(query, Wq, bq, 0.125f, (void*)Qb);
  mha_proj_wmma<0><<<gproj, blk32, 0, stream>>>(value, Wk, bk, 1.0f,   (void*)Kb);
  mha_proj_wmma<1><<<gproj, blk32, 0, stream>>>(value, Wv, bv, 1.0f,   (void*)Vt);

  const dim3 gfa(32, 16, 2);  // (T/64, H, B), 128 threads
  mha_flash_wmma<<<gfa, dim3(128), 0, stream>>>(Qb, Kb, Vt, amask, ctx);

  mha_proj_wmma<2><<<gproj, blk32, 0, stream>>>(ctx, Wo, bo, 1.0f, (void*)out);
}
